// Model_65455301591209
// MI455X (gfx1250) — matrix operand order repaired and checked by reading, not compiled, not run on hardware
//
#include <hip/hip_runtime.h>
#include <hip/hip_fp16.h>
#include <math.h>

// ---------- model dims ----------
#define BB   8
#define TT   512
#define DM   512      // D_MODEL (== conv in-channels == T)
#define LT   5        // L_TCN
#define NT   1000     // NUM_TOK
#define NTP  1008     // NT padded to 16-row tiles (A operand)
#define NTW  1024     // NT padded to 64-row tiles (W operand)
#define VOC  32000
#define DL   4096     // D_LLM
#define H8   8        // cross-attn heads
#define EE   128      // D_FF per head
#define DKn  1024     // D_FF * N_HEADS
#define KNT  1024     // cross-attn PV K (1000) padded to mult-of-32
#define SS   133      // prompt 128 + 5
#define SPD  144      // S padded to 16-row tiles (A operand)
#define SP2  192      // S padded to 64-row tiles (W operand: q/k heads)
#define SPK  160      // S padded to mult-of-32 (PV K dim)
#define HHn  32       // llama heads
#define HDn  128      // llama head dim
#define FFNn 11008
#define NLn  2
#define MSR  1064     // B*S token rows
#define MSP  1072     // padded token rows

typedef __attribute__((ext_vector_type(16))) _Float16 v16h;
typedef __attribute__((ext_vector_type(8)))  _Float16 v8h;
typedef __attribute__((ext_vector_type(8)))  float    v8f;

// ---------- branch-free fp16 fragment loaders (wave32 WMMA layouts) ----------
// A: 16x32 f16. lane m = m0+(lane&15); halves = K kb..kb+15,
// kb = k0 + 16*(lane>>4). Buffers padded so all loads are in-bounds & 16B aligned.
__device__ __forceinline__ v16h load_a_frag(const _Float16* __restrict__ A, long long base,
                                            int lda, int m0, int k0, int lane) {
    int m  = m0 + (lane & 15);
    int kb = k0 + ((lane >> 4) << 4);
    const _Float16* p = A + base + (long long)m * lda + kb;
    v8h lo = *(const v8h*)p;
    v8h hi = *(const v8h*)(p + 8);
    v16h r;
    #pragma unroll
    for (int i = 0; i < 8; ++i) { r[i] = lo[i]; r[8 + i] = hi[i]; }
    return r;
}
// B = W^T (W row-major NxK): lane col n = n0+(lane&15); halves = K kb..kb+15,
// kb = k0 + 16*(lane>>4). Two adjacent 16B loads.
__device__ __forceinline__ v16h load_w_frag(const _Float16* __restrict__ W, long long base,
                                            int ldw, int n0, int k0, int lane) {
    int n  = n0 + (lane & 15);
    int kb = k0 + ((lane >> 4) << 4);
    const _Float16* p = W + base + (long long)n * ldw + kb;
    v8h lo = *(const v8h*)p;
    v8h hi = *(const v8h*)(p + 8);
    v16h r;
    #pragma unroll
    for (int i = 0; i < 8; ++i) { r[i] = lo[i]; r[8 + i] = hi[i]; }
    return r;
}

// ---------- generic batched WMMA GEMM: C[z] = A[z] * W[z]^T (+bias) ----------
// Each wave computes a 16x64 tile: one A fragment reused across 4 WMMAs per K-step.
// Requires: A rows padded to 16-tile (wave-uniform m0 guard), W rows padded to 64-tile.
// z = blockIdx.z; (zo,zi) = (z/zdiv, z%zdiv); per-matrix offset = zo*Outer + zi*Inner.
// biasMode: 0=none, 1=per-column bias[n], 2=per-row bias[m].
__global__ void k_gemm(const _Float16* __restrict__ A, const _Float16* __restrict__ W,
                       float* __restrict__ C, const float* __restrict__ bias, int biasMode,
                       int M, int N, int K, int lda, int ldw, int ldc, int zdiv,
                       long long aO, long long aI, long long wO, long long wI,
                       long long cO, long long cI) {
    int lane = threadIdx.x;
    int n0 = blockIdx.x * 64;
    int m0 = (blockIdx.y * blockDim.y + threadIdx.y) * 16;
    if (m0 >= M) return;                       // wave-uniform: EXEC all-ones at WMMA
    int z  = blockIdx.z;
    int zo = z / zdiv, zi = z % zdiv;
    long long ab = (long long)zo * aO + (long long)zi * aI;
    long long wb = (long long)zo * wO + (long long)zi * wI;
    long long cb = (long long)zo * cO + (long long)zi * cI;

    v8f acc0 = {}, acc1 = {}, acc2 = {}, acc3 = {};
    for (int k0 = 0; k0 < K; k0 += 32) {
        v16h a  = load_a_frag(A, ab, lda, m0, k0, lane);
        v16h w0 = load_w_frag(W, wb, ldw, n0,      k0, lane);
        v16h w1 = load_w_frag(W, wb, ldw, n0 + 16, k0, lane);
        v16h w2 = load_w_frag(W, wb, ldw, n0 + 32, k0, lane);
        v16h w3 = load_w_frag(W, wb, ldw, n0 + 48, k0, lane);
        acc0 = __builtin_amdgcn_wmma_f32_16x16x32_f16(false, a, false, w0, (short)0, acc0, false, false);
        acc1 = __builtin_amdgcn_wmma_f32_16x16x32_f16(false, a, false, w1, (short)0, acc1, false, false);
        acc2 = __builtin_amdgcn_wmma_f32_16x16x32_f16(false, a, false, w2, (short)0, acc2, false, false);
        acc3 = __builtin_amdgcn_wmma_f32_16x16x32_f16(false, a, false, w3, (short)0, acc3, false, false);
    }
    int mB = m0 + ((lane >> 4) << 3);
    v8f accs[4] = {acc0, acc1, acc2, acc3};
    #pragma unroll
    for (int j = 0; j < 4; ++j) {
        int n = n0 + j * 16 + (lane & 15);
        if (n < N) {
            float bn = (biasMode == 1) ? bias[n] : 0.f;
            #pragma unroll
            for (int v = 0; v < 8; ++v) {
                int m = mB + v;
                if (m < M) {
                    float bm = (biasMode == 2) ? bias[m] : 0.f;
                    C[cb + (long long)m * ldc + n] = accs[j][v] + bn + bm;
                }
            }
        }
    }
}

// ---------- small kernels ----------
__global__ void k_f32_to_f16(const float* __restrict__ s, _Float16* __restrict__ d, long long n) {
    long long i = (long long)blockIdx.x * blockDim.x + threadIdx.x;
    if (i < n) d[i] = (_Float16)s[i];
}
__global__ void k_fill16(_Float16* __restrict__ d, long long n) {
    long long i = (long long)blockIdx.x * blockDim.x + threadIdx.x;
    if (i < n) d[i] = (_Float16)0.f;
}
__global__ void k_f32_to_f16_T(const float* __restrict__ s, _Float16* __restrict__ d, int R, int Cc) {
    long long i = (long long)blockIdx.x * blockDim.x + threadIdx.x;
    if (i < (long long)R * Cc) {
        int r = (int)(i / Cc), c = (int)(i % Cc);
        d[(long long)c * R + r] = (_Float16)s[i];
    }
}
__global__ void k_revin(const float* __restrict__ x, float* mean, float* stdv) {
    __shared__ float s1[256], s2[256];
    int b = blockIdx.x, t = threadIdx.x;
    float a = 0.f, q = 0.f;
    for (int i = t; i < TT; i += 256) { float v = x[b * TT + i]; a += v; q += v * v; }
    s1[t] = a; s2[t] = q; __syncthreads();
    for (int o = 128; o > 0; o >>= 1) { if (t < o) { s1[t] += s1[t + o]; s2[t] += s2[t + o]; } __syncthreads(); }
    if (t == 0) { float m = s1[0] / TT; float v = s2[0] / TT - m * m; mean[b] = m; stdv[b] = sqrtf(v + 1e-5f); }
}
__global__ void k_conv1(const float* __restrict__ x, const float* mean, const float* stdv,
                        const float* __restrict__ w, const float* __restrict__ bias, float* __restrict__ y) {
    int idx = blockIdx.x * blockDim.x + threadIdx.x;
    if (idx >= BB * DM * 3) return;
    int h = idx % 3, o = (idx / 3) % DM, b = idx / (3 * DM);
    int k = 2 - h;
    float m = mean[b], sd = stdv[b], acc = bias[o];
    for (int c = 0; c < DM; ++c)
        acc += w[(o * DM + c) * 3 + k] * ((x[b * TT + c] - m) / sd);
    y[idx] = fmaxf(acc, 0.f);                      // relu, layout (B, DM, 3)
}
// conv2 -> enc16 (48 padded rows x 512): rows 0..39 = (b,l), pad rows zeroed by fill.
__global__ void k_conv2(const float* __restrict__ y1, const float* __restrict__ w,
                        const float* __restrict__ bias, _Float16* __restrict__ enc16) {
    int idx = blockIdx.x * blockDim.x + threadIdx.x;
    if (idx >= BB * DM * LT) return;
    int h = idx % LT, o = (idx / LT) % DM, b = idx / (LT * DM);
    float acc = bias[o];
    for (int c = 0; c < DM; ++c)
        #pragma unroll
        for (int k = 0; k < 3; ++k) {
            int hi = h + k - 2;
            if (hi >= 0 && hi < 3) acc += w[(o * DM + c) * 3 + k] * y1[(b * DM + c) * 3 + hi];
        }
    enc16[((long long)b * LT + h) * DM + o] = (_Float16)acc;
}
// cross softmax: scores (z=b*8+h batches, 5 x 1000, ld 1000) -> P16 (z, 16, 1024) zero-padded
__global__ void k_softmax_cross(const float* __restrict__ S, _Float16* __restrict__ P,
                                const float* alpha, const float* beta) {
    __shared__ float red[256];
    int r = blockIdx.x, t = threadIdx.x;
    int z = r / 16, l = r % 16;
    _Float16* prow = P + (long long)r * KNT;
    if (l >= LT) {
        for (int i = t; i < KNT; i += 256) prow[i] = (_Float16)0.f;
        return;
    }
    float scale = alpha[0] * beta[0] * 0.08838834764831845f;   // ab / sqrt(128)
    const float* row = S + ((long long)z * LT + l) * NT;
    float mx = -1e30f;
    for (int i = t; i < NT; i += 256) mx = fmaxf(mx, row[i] * scale);
    red[t] = mx; __syncthreads();
    for (int o = 128; o > 0; o >>= 1) { if (t < o) red[t] = fmaxf(red[t], red[t + o]); __syncthreads(); }
    mx = red[0]; __syncthreads();
    float sm = 0.f;
    for (int i = t; i < NT; i += 256) sm += expf(row[i] * scale - mx);
    red[t] = sm; __syncthreads();
    for (int o = 128; o > 0; o >>= 1) { if (t < o) red[t] += red[t + o]; __syncthreads(); }
    float inv = 1.f / red[0];
    for (int i = t; i < KNT; i += 256)
        prow[i] = (i < NT) ? (_Float16)(expf(row[i] * scale - mx) * inv) : (_Float16)0.f;
}
// causal softmax: scores (z=b*32+h, 133x133 ld 133) -> P16 (z, 144, 160) zero-padded
__global__ void k_softmax_causal(const float* __restrict__ S, _Float16* __restrict__ P) {
    __shared__ float red[256];
    int r = blockIdx.x, t = threadIdx.x;
    int z = r / SPD, q = r % SPD;
    _Float16* prow = P + (long long)r * SPK;
    if (q >= SS) {
        for (int i = t; i < SPK; i += 256) prow[i] = (_Float16)0.f;
        return;
    }
    const float scale = 0.08838834764831845f;
    const float* row = S + ((long long)z * SS + q) * SS;
    int len = q + 1;
    float mx = -1e30f;
    for (int i = t; i < len; i += 256) mx = fmaxf(mx, row[i] * scale);
    red[t] = mx; __syncthreads();
    for (int o = 128; o > 0; o >>= 1) { if (t < o) red[t] = fmaxf(red[t], red[t + o]); __syncthreads(); }
    mx = red[0]; __syncthreads();
    float sm = 0.f;
    for (int i = t; i < len; i += 256) sm += expf(row[i] * scale - mx);
    red[t] = sm; __syncthreads();
    for (int o = 128; o > 0; o >>= 1) { if (t < o) red[t] += red[t + o]; __syncthreads(); }
    float inv = 1.f / red[0];
    for (int i = t; i < SPK; i += 256)
        prow[i] = (i < len) ? (_Float16)(expf(row[i] * scale - mx) * inv) : (_Float16)0.f;
}
// RoPE: X (B, 133, 32, 128) f32 -> Y (B, 192, 32, 128) f16, pad rows written as zeros
__global__ void k_rope(const float* __restrict__ X, _Float16* __restrict__ Y) {
    long long i = (long long)blockIdx.x * blockDim.x + threadIdx.x;
    long long total = (long long)BB * SP2 * HHn * (HDn / 2);
    if (i >= total) return;
    int d = (int)(i % 64); long long r = i / 64;
    int hh = (int)(r % HHn); r /= HHn;
    int s = (int)(r % SP2);  int b = (int)(r / SP2);
    long long obase = (((long long)b * SP2 + s) * HHn + hh) * HDn;
    if (s >= SS) {
        Y[obase + d] = (_Float16)0.f;
        Y[obase + 64 + d] = (_Float16)0.f;
        return;
    }
    float inv = powf(10000.f, -(2.f * d) / (float)HDn);
    float ang = s * inv, c = cosf(ang), sn = sinf(ang);
    long long ibase = (((long long)b * SS + s) * HHn + hh) * HDn;
    float x1 = X[ibase + d], x2 = X[ibase + 64 + d];
    Y[obase + d]      = (_Float16)(x1 * c - x2 * sn);
    Y[obase + 64 + d] = (_Float16)(x1 * sn + x2 * c);
}
// V (B,133,32,128) f32 -> Vt (B,32,128,160) f16 zero-padded along s
__global__ void k_vtrans(const float* __restrict__ V, _Float16* __restrict__ Vt) {
    long long i = (long long)blockIdx.x * blockDim.x + threadIdx.x;
    long long total = (long long)BB * HHn * HDn * SPK;
    if (i >= total) return;
    int s = (int)(i % SPK); long long r = i / SPK;
    int d = (int)(r % HDn); r /= HDn;
    int hh = (int)(r % HHn); int b = (int)(r / HHn);
    float v = (s < SS) ? V[(((long long)b * SS + s) * HHn + hh) * HDn + d] : 0.f;
    Vt[i] = (_Float16)v;
}
// V (1000,8,128) f32 -> Vt (8,128,1024) f16 zero-padded along s
__global__ void k_vtrans_cross(const float* __restrict__ V, _Float16* __restrict__ Vt) {
    long long i = (long long)blockIdx.x * blockDim.x + threadIdx.x;
    if (i >= (long long)H8 * EE * KNT) return;
    int s = (int)(i % KNT); long long r = i / KNT;
    int e = (int)(r % EE); int h = (int)(r / EE);
    Vt[i] = (s < NT) ? (_Float16)V[((long long)s * H8 + h) * EE + e] : (_Float16)0.f;
}
__global__ void k_rmsnorm16(const float* __restrict__ X, const float* __restrict__ g,
                            _Float16* __restrict__ Y, int D) {
    __shared__ float red[256];
    __shared__ float invr;
    int r = blockIdx.x, t = threadIdx.x;
    const float* row = X + (long long)r * D;
    float s = 0.f;
    for (int i = t; i < D; i += 256) { float v = row[i]; s += v * v; }
    red[t] = s; __syncthreads();
    for (int o = 128; o > 0; o >>= 1) { if (t < o) red[t] += red[t + o]; __syncthreads(); }
    if (t == 0) invr = rsqrtf(red[0] / D + 1e-6f);
    __syncthreads();
    for (int i = t; i < D; i += 256) Y[(long long)r * D + i] = (_Float16)(row[i] * invr * g[i]);
}
__global__ void k_rmsnorm32(const float* __restrict__ X, const float* __restrict__ g,
                            float* __restrict__ Y, int D) {
    __shared__ float red[256];
    __shared__ float invr;
    int r = blockIdx.x, t = threadIdx.x;
    const float* row = X + (long long)r * D;
    float s = 0.f;
    for (int i = t; i < D; i += 256) { float v = row[i]; s += v * v; }
    red[t] = s; __syncthreads();
    for (int o = 128; o > 0; o >>= 1) { if (t < o) red[t] += red[t + o]; __syncthreads(); }
    if (t == 0) invr = rsqrtf(red[0] / D + 1e-6f);
    __syncthreads();
    for (int i = t; i < D; i += 256) Y[(long long)r * D + i] = row[i] * invr * g[i];
}
__global__ void k_silu_mul(const float* __restrict__ G, const float* __restrict__ U,
                           _Float16* __restrict__ Y, long long n) {
    long long i = (long long)blockIdx.x * blockDim.x + threadIdx.x;
    if (i < n) { float g = G[i]; Y[i] = (_Float16)((g / (1.f + expf(-g))) * U[i]); }
}
__global__ void k_add(float* __restrict__ H, const float* __restrict__ X, long long n) {
    long long i = (long long)blockIdx.x * blockDim.x + threadIdx.x;
    if (i < n) H[i] += X[i];
}
__global__ void k_concat(const float* __restrict__ prompt, const float* __restrict__ enc,
                         float* __restrict__ H) {
    long long i = (long long)blockIdx.x * blockDim.x + threadIdx.x;
    if (i >= (long long)BB * SS * DL) return;
    int d = (int)(i % DL);
    int s = (int)((i / DL) % SS);
    int b = (int)(i / ((long long)SS * DL));
    H[i] = (s < 128) ? prompt[((long long)b * 128 + s) * DL + d]
                     : enc[((long long)b * LT + (s - 128)) * DL + d];
}
__global__ void k_head(const float* __restrict__ HN, const float* __restrict__ hw,
                       const float* __restrict__ hb, const float* mean, const float* stdv,
                       float* __restrict__ out) {
    __shared__ float red[256];
    int bt = blockIdx.x, b = bt / 16, tt = bt % 16, t = threadIdx.x;
    float s = 0.f;
    for (int j = t; j < 8192; j += 256) {
        int e = j / 64, p = j % 64;
        s += HN[((long long)b * SS + (69 + p)) * DL + e] * hw[tt * 8192 + j];
    }
    red[t] = s; __syncthreads();
    for (int o = 128; o > 0; o >>= 1) { if (t < o) red[t] += red[t + o]; __syncthreads(); }
    if (t == 0) out[b * 16 + tt] = (red[0] + hb[tt]) * stdv[b] + mean[b];
}

// ---------- host ----------
static inline size_t al256(size_t x) { return (x + 255) & ~(size_t)255; }

extern "C" void kernel_launch(void* const* d_in, const int* in_sizes, int n_in,
                              void* d_out, int out_size, void* d_ws, size_t ws_size,
                              hipStream_t stream) {
    const float* x_enc   = (const float*)d_in[0];
    const float* prompt  = (const float*)d_in[1];
    const float* wordemb = (const float*)d_in[2];
    const float* map_w   = (const float*)d_in[3];
    const float* map_b   = (const float*)d_in[4];
    const float* c1w = (const float*)d_in[5];  const float* c1b = (const float*)d_in[6];
    const float* c2w = (const float*)d_in[7];  const float* c2b = (const float*)d_in[8];
    const float* q_w = (const float*)d_in[9];  const float* q_b = (const float*)d_in[10];
    const float* k_w = (const float*)d_in[11]; const float* k_b = (const float*)d_in[12];
    const float* v_w = (const float*)d_in[13]; const float* v_b = (const float*)d_in[14];
    const float* o_w = (const float*)d_in[15]; const float* o_b = (const float*)d_in[16];
    const float* alpha = (const float*)d_in[17];
    const float* beta  = (const float*)d_in[18];
    const float* wq = (const float*)d_in[19]; const float* wk = (const float*)d_in[20];
    const float* wv = (const float*)d_in[21]; const float* wo = (const float*)d_in[22];
    const float* wg = (const float*)d_in[23]; const float* wu = (const float*)d_in[24];
    const float* wd = (const float*)d_in[25];
    const float* ln1 = (const float*)d_in[26]; const float* ln2 = (const float*)d_in[27];
    const float* fln = (const float*)d_in[28];
    const float* hw  = (const float*)d_in[29]; const float* hb = (const float*)d_in[30];
    float* outp = (float*)d_out;

    // bump allocator over d_ws
    char* base = (char*)d_ws;
    size_t off = 0;
    auto A = [&](size_t bytes) { size_t o = off; off = al256(off + bytes); return (void*)(base + o); };

    float*    mean    = (float*)A(BB * 4);
    float*    stdv    = (float*)A(BB * 4);
    _Float16* we16T   = (_Float16*)A((size_t)DL * VOC * 2);
    _Float16* mapw16  = (_Float16*)A((size_t)NTP * VOC * 2);         // A rows padded 1008
    float*    src32   = (float*)A((size_t)NT * DL * 4);
    _Float16* src16   = (_Float16*)A((size_t)NTP * DL * 2);          // A rows padded 1008
    float*    y1      = (float*)A((size_t)BB * DM * 3 * 4);
    _Float16* enc16   = (_Float16*)A((size_t)48 * DM * 2);           // A rows padded 48
    _Float16* qw16    = (_Float16*)A((size_t)DKn * DM * 2);
    _Float16* kw16    = (_Float16*)A((size_t)DKn * DL * 2);
    _Float16* vw16    = (_Float16*)A((size_t)DKn * DL * 2);
    _Float16* ow16    = (_Float16*)A((size_t)DL * DKn * 2);
    float*    q32c    = (float*)A((size_t)BB * LT * DKn * 4);
    float*    k32c    = (float*)A((size_t)NT * DKn * 4);
    float*    v32c    = (float*)A((size_t)NT * DKn * 4);
    _Float16* q16c    = (_Float16*)A((size_t)64 * DKn * 2);          // A rows padded 64 (tile reach 51)
    _Float16* k16c    = (_Float16*)A((size_t)NTW * DKn * 2);         // W rows padded 1024
    _Float16* vt16c   = (_Float16*)A((size_t)H8 * EE * KNT * 2);     // K padded 1024
    float*    sc32c   = (float*)A((size_t)BB * H8 * LT * NT * 4);
    _Float16* p16c    = (_Float16*)A((size_t)BB * H8 * 16 * KNT * 2);// 16 rows/batch, K pad 1024
    float*    rep32   = (float*)A((size_t)BB * LT * DKn * 4);
    _Float16* rep16   = (_Float16*)A((size_t)48 * DKn * 2);          // A rows padded 48
    float*    encA32  = (float*)A((size_t)BB * LT * DL * 4);
    float*    h32     = (float*)A((size_t)MSR * DL * 4);
    _Float16* wq16    = (_Float16*)A((size_t)NLn * DL * DL * 2);
    _Float16* wk16    = (_Float16*)A((size_t)NLn * DL * DL * 2);
    _Float16* wv16    = (_Float16*)A((size_t)NLn * DL * DL * 2);
    _Float16* wo16    = (_Float16*)A((size_t)NLn * DL * DL * 2);
    _Float16* wg16    = (_Float16*)A((size_t)NLn * FFNn * DL * 2);
    _Float16* wu16    = (_Float16*)A((size_t)NLn * FFNn * DL * 2);
    _Float16* wd16    = (_Float16*)A((size_t)NLn * DL * FFNn * 2);
    _Float16* x16     = (_Float16*)A((size_t)MSP * DL * 2);          // A rows padded 1072
    float*    q32l    = (float*)A((size_t)MSR * DL * 4);
    float*    k32l    = (float*)A((size_t)MSR * DL * 4);
    float*    v32l    = (float*)A((size_t)MSR * DL * 4);
    _Float16* qh16    = (_Float16*)A((size_t)BB * SP2 * DL * 2);     // (B,192,32,128)
    _Float16* kh16    = (_Float16*)A((size_t)BB * SP2 * DL * 2);     // (B,192,32,128)
    _Float16* vt16l   = (_Float16*)A((size_t)BB * HHn * HDn * SPK * 2);
    float*    sc32l   = (float*)A((size_t)BB * HHn * SS * SS * 4);
    _Float16* p16l    = (_Float16*)A((size_t)BB * HHn * SPD * SPK * 2);
    float*    attn32  = (float*)A((size_t)MSR * DL * 4);
    _Float16* attn16  = (_Float16*)A((size_t)MSP * DL * 2);          // A rows padded 1072
    float*    proj32  = (float*)A((size_t)MSR * DL * 4);
    float*    g32     = (float*)A((size_t)MSR * FFNn * 4);
    float*    u32     = (float*)A((size_t)MSR * FFNn * 4);
    _Float16* ff16    = (_Float16*)A((size_t)MSP * FFNn * 2);        // A rows padded 1072
    float*    down32  = (float*)A((size_t)MSR * DL * 4);
    float*    hn32    = (float*)A((size_t)MSR * DL * 4);
    (void)ws_size; (void)n_in; (void)in_sizes; (void)out_size;

    auto cvt = [&](const float* s, _Float16* d, long long n) {
        k_f32_to_f16<<<(unsigned)((n + 255) / 256), 256, 0, stream>>>(s, d, n);
    };
    auto fill = [&](_Float16* d, long long n) {
        k_fill16<<<(unsigned)((n + 255) / 256), 256, 0, stream>>>(d, n);
    };
    auto gemm = [&](const _Float16* Am, const _Float16* Wm, float* Cm, const float* bias,
                    int biasMode, int M, int N, int K, int lda, int ldw, int ldc,
                    int batch, int zdiv, long long aO, long long aI, long long wO,
                    long long wI, long long cO, long long cI) {
        dim3 blk(32, 4);
        dim3 grd((N + 63) / 64, (M + 63) / 64, batch);
        k_gemm<<<grd, blk, 0, stream>>>(Am, Wm, Cm, bias, biasMode, M, N, K, lda, ldw, ldc,
                                        zdiv, aO, aI, wO, wI, cO, cI);
    };

    // ---- weight conversions (fp32 -> fp16, word_emb transposed to (DL, VOC)) ----
    {
        long long n = (long long)VOC * DL;
        k_f32_to_f16_T<<<(unsigned)((n + 255) / 256), 256, 0, stream>>>(wordemb, we16T, VOC, DL);
    }
    cvt(map_w, mapw16, (long long)NT * VOC);
    cvt(q_w, qw16, (long long)DKn * DM);
    cvt(k_w, kw16, (long long)DKn * DL);
    cvt(v_w, vw16, (long long)DKn * DL);
    cvt(o_w, ow16, (long long)DL * DKn);
    cvt(wq, wq16, (long long)NLn * DL * DL);
    cvt(wk, wk16, (long long)NLn * DL * DL);
    cvt(wv, wv16, (long long)NLn * DL * DL);
    cvt(wo, wo16, (long long)NLn * DL * DL);
    cvt(wg, wg16, (long long)NLn * FFNn * DL);
    cvt(wu, wu16, (long long)NLn * FFNn * DL);
    cvt(wd, wd16, (long long)NLn * DL * FFNn);

    // ---- zero the pad rows of every padded fp16 operand (read by unguarded loads) ----
    fill(mapw16 + (size_t)NT * VOC, (long long)(NTP - NT) * VOC);
    fill(src16  + (size_t)NT * DL,  (long long)(NTP - NT) * DL);
    fill(enc16  + (size_t)40 * DM,  (long long)8 * DM);
    fill(q16c   + (size_t)40 * DKn, (long long)24 * DKn);
    fill(k16c   + (size_t)NT * DKn, (long long)(NTW - NT) * DKn);
    fill(rep16  + (size_t)40 * DKn, (long long)8 * DKn);
    fill(x16    + (size_t)MSR * DL, (long long)(MSP - MSR) * DL);
    fill(attn16 + (size_t)MSR * DL, (long long)(MSP - MSR) * DL);
    fill(ff16   + (size_t)MSR * FFNn, (long long)(MSP - MSR) * FFNn);

    // ---- RevIN + TCN convs -> enc16 ----
    k_revin<<<BB, 256, 0, stream>>>(x_enc, mean, stdv);
    k_conv1<<<(BB * DM * 3 + 255) / 256, 256, 0, stream>>>(x_enc, mean, stdv, c1w, c1b, y1);
    k_conv2<<<(BB * DM * LT + 255) / 256, 256, 0, stream>>>(y1, c2w, c2b, enc16);

    // ---- src = map_w @ word_emb + map_b[:,None]   (1000x4096, K=32000) ----
    gemm(mapw16, we16T, src32, map_b, 2, NT, DL, VOC, VOC, VOC, DL, 1, 1, 0, 0, 0, 0, 0, 0);
    cvt(src32, src16, (long long)NT * DL);

    // ---- cross-attn projections ----
    gemm(enc16, qw16, q32c, q_b, 1, BB * LT, DKn, DM, DM, DM, DKn, 1, 1, 0, 0, 0, 0, 0, 0);
    gemm(src16, kw16, k32c, k_b, 1, NT, DKn, DL, DL, DL, DKn, 1, 1, 0, 0, 0, 0, 0, 0);
    gemm(src16, vw16, v32c, v_b, 1, NT, DKn, DL, DL, DL, DKn, 1, 1, 0, 0, 0, 0, 0, 0);
    cvt(q32c, q16c, (long long)BB * LT * DKn);
    cvt(k32c, k16c, (long long)NT * DKn);
    {
        long long n = (long long)H8 * EE * KNT;
        k_vtrans_cross<<<(unsigned)((n + 255) / 256), 256, 0, stream>>>(v32c, vt16c);
    }
    // scores[b,h,l,s] : batch z=b*8+h, K=128 (W rows padded to 1024)
    gemm(q16c, k16c, sc32c, nullptr, 0, LT, NT, EE, DKn, DKn, NT,
         BB * H8, H8, (long long)LT * DKn, EE, 0, EE,
         (long long)H8 * LT * NT, (long long)LT * NT);
    k_softmax_cross<<<BB * H8 * 16, 256, 0, stream>>>(sc32c, p16c, alpha, beta);
    // rep[b,l,h,e] : P(16x1024 padded) x Vt(128x1024)^T, K=1024 (zero-padded from 1000)
    gemm(p16c, vt16c, rep32, nullptr, 0, LT, EE, KNT, KNT, KNT, DKn,
         BB * H8, H8, (long long)H8 * 16 * KNT, (long long)16 * KNT,
         0, (long long)EE * KNT, (long long)LT * DKn, EE);
    cvt(rep32, rep16, (long long)BB * LT * DKn);
    gemm(rep16, ow16, encA32, o_b, 1, BB * LT, DL, DKn, DKn, DKn, DL, 1, 1, 0, 0, 0, 0, 0, 0);

    // ---- concat prompt + enc -> h32 (B,133,4096) ----
    {
        long long n = (long long)MSR * DL;
        k_concat<<<(unsigned)((n + 255) / 256), 256, 0, stream>>>(prompt, encA32, h32);
    }

    // ---- LLaMA layers ----
    for (int i = 0; i < NLn; ++i) {
        const _Float16* Wq = wq16 + (size_t)i * DL * DL;
        const _Float16* Wk = wk16 + (size_t)i * DL * DL;
        const _Float16* Wv = wv16 + (size_t)i * DL * DL;
        const _Float16* Wo = wo16 + (size_t)i * DL * DL;
        const _Float16* Wg = wg16 + (size_t)i * FFNn * DL;
        const _Float16* Wu = wu16 + (size_t)i * FFNn * DL;
        const _Float16* Wd = wd16 + (size_t)i * DL * FFNn;
        const float* g1 = ln1 + (size_t)i * DL;
        const float* g2 = ln2 + (size_t)i * DL;
        long long nBD = (long long)MSR * DL;

        k_rmsnorm16<<<MSR, 256, 0, stream>>>(h32, g1, x16, DL);
        gemm(x16, Wq, q32l, nullptr, 0, MSR, DL, DL, DL, DL, DL, 1, 1, 0, 0, 0, 0, 0, 0);
        gemm(x16, Wk, k32l, nullptr, 0, MSR, DL, DL, DL, DL, DL, 1, 1, 0, 0, 0, 0, 0, 0);
        gemm(x16, Wv, v32l, nullptr, 0, MSR, DL, DL, DL, DL, DL, 1, 1, 0, 0, 0, 0, 0, 0);
        {
            long long n = (long long)BB * SP2 * HHn * 64;
            k_rope<<<(unsigned)((n + 255) / 256), 256, 0, stream>>>(q32l, qh16);
            k_rope<<<(unsigned)((n + 255) / 256), 256, 0, stream>>>(k32l, kh16);
            long long nv = (long long)BB * HHn * HDn * SPK;
            k_vtrans<<<(unsigned)((nv + 255) / 256), 256, 0, stream>>>(v32l, vt16l);
        }
        // scores[b,h,q,k] : batch z = b*32+h, K=HD=128; A rows pad 144-reach, W rows pad 192
        gemm(qh16, kh16, sc32l, nullptr, 0, SS, SS, HDn, DL, DL, SS,
             BB * HHn, HHn, (long long)SP2 * DL, HDn, (long long)SP2 * DL, HDn,
             (long long)HHn * SS * SS, (long long)SS * SS);
        k_softmax_causal<<<BB * HHn * SPD, 256, 0, stream>>>(sc32l, p16l);
        // attn[b,q,h,d] : P(144x160) x Vt(128x160)^T, K=160 (zero-padded from 133)
        gemm(p16l, vt16l, attn32, nullptr, 0, SS, HDn, SPK, SPK, SPK, DL,
             BB * HHn, HHn, (long long)HHn * SPD * SPK, (long long)SPD * SPK,
             (long long)HHn * HDn * SPK, (long long)HDn * SPK,
             (long long)SS * DL, HDn);
        cvt(attn32, attn16, nBD);
        gemm(attn16, Wo, proj32, nullptr, 0, MSR, DL, DL, DL, DL, DL, 1, 1, 0, 0, 0, 0, 0, 0);
        k_add<<<(unsigned)((nBD + 255) / 256), 256, 0, stream>>>(h32, proj32, nBD);

        k_rmsnorm16<<<MSR, 256, 0, stream>>>(h32, g2, x16, DL);
        gemm(x16, Wg, g32, nullptr, 0, MSR, FFNn, DL, DL, DL, FFNn, 1, 1, 0, 0, 0, 0, 0, 0);
        gemm(x16, Wu, u32, nullptr, 0, MSR, FFNn, DL, DL, DL, FFNn, 1, 1, 0, 0, 0, 0, 0, 0);
        {
            long long n = (long long)MSR * FFNn;
            k_silu_mul<<<(unsigned)((n + 255) / 256), 256, 0, stream>>>(g32, u32, ff16, n);
        }
        gemm(ff16, Wd, down32, nullptr, 0, MSR, DL, FFNn, FFNn, FFNn, DL, 1, 1, 0, 0, 0, 0, 0, 0);
        k_add<<<(unsigned)((nBD + 255) / 256), 256, 0, stream>>>(h32, down32, nBD);
    }

    // ---- final norm + head + RevIN denorm ----
    k_rmsnorm32<<<MSR, 256, 0, stream>>>(h32, fln, hn32, DL);
    k_head<<<BB * 16, 256, 0, stream>>>(hn32, hw, hb, mean, stdv, outp);
}
